// Attention_66030827208983
// MI455X (gfx1250) — compile-verified, matrix operand order checked by reading, not run on hardware
//
#include <hip/hip_runtime.h>
#include <hip/hip_bf16.h>
#include <math.h>

typedef __attribute__((ext_vector_type(16))) _Float16 v16h;
typedef __attribute__((ext_vector_type(8)))  float    v8f;

#define DIM   768
#define NTOK  4096
#define NB    4
#define QKVC  2304   // 3*DIM
#define KB    64     // keys per attention block iteration

// ---------------- TDM availability ----------------
#if defined(__has_builtin)
#if __has_builtin(__builtin_amdgcn_tensor_load_to_lds)
#define HAVE_TDM 1
#endif
#endif
#ifndef HAVE_TDM
#define HAVE_TDM 0
#endif

#if HAVE_TDM
typedef unsigned int u32x4 __attribute__((ext_vector_type(4)));
typedef int          i32x4 __attribute__((ext_vector_type(4)));
typedef int          i32x8 __attribute__((ext_vector_type(8)));

__device__ __forceinline__ unsigned attn66_lds_offset(const void* p) {
    // generic -> LDS address space; ptrtoint of an AS(3) pointer yields the 32-bit LDS offset
    return (unsigned)(unsigned long long)(__attribute__((address_space(3))) const void*)p;
}

// 2D f16 tile DMA: tile_d0 contiguous elements per row, tile_d1 rows, row stride = stride0
// elements. Descriptor bitfields per CDNA5 ISA 8.3/8.4 (D# group0/group1).
__device__ __forceinline__ void attn66_tdm_load_2d_f16(unsigned lds_addr, const void* gaddr,
                                                       unsigned tensor_d0, unsigned tensor_d1,
                                                       unsigned tile_d0, unsigned tile_d1,
                                                       unsigned long long stride0) {
    unsigned long long ga = (unsigned long long)gaddr;
    u32x4 g0 = {
        1u,                                                   // count=1, user descriptor
        lds_addr,                                             // lds_addr [63:32]
        (unsigned)(ga & 0xffffffffull),                       // global_addr[31:0]
        (unsigned)((ga >> 32) & 0x1ffffffull) | (2u << 30)    // global_addr[56:32] | type=2
    };
    i32x8 g1 = {
        (int)(1u << 16),                                              // data_size=1 -> 2 bytes
        (int)((tensor_d0 & 0xffffu) << 16),                           // tensor_dim0[15:0] @bit48
        (int)(((tensor_d0 >> 16) & 0xffffu) | ((tensor_d1 & 0xffffu) << 16)),
        (int)(((tensor_d1 >> 16) & 0xffffu) | ((tile_d0 & 0xffffu) << 16)),
        (int)(tile_d1 & 0xffffu),                                     // tile_dim1 (tile_dim2=0)
        (int)(unsigned)(stride0 & 0xffffffffull),                     // dim0_stride[31:0]
        (int)(unsigned)((stride0 >> 32) & 0xffffull),                 // dim0_stride[47:32]
        0
    };
    i32x4 z4 = {0, 0, 0, 0};
    i32x8 z8 = {0, 0, 0, 0, 0, 0, 0, 0};
    // clang-23 / therock-10.0 6-arg form: (g0, g1, g2, g3, extra, cpol)
    __builtin_amdgcn_tensor_load_to_lds(g0, g1, z4, z4, z8, 0);
}
#endif

// Load a 16-bit WMMA A/B fragment from row-major storage.
// Caller passes pointer to element [m_or_n = lane&15][bk = (lane>>4)*8].
// Halves 0..7  <- K = bk..bk+7   (16 bytes at p)
// Halves 8..15 <- K = bk+16..+23 (16 bytes at p+16)
__device__ __forceinline__ v16h load_frag16(const _Float16* p) {
    v16h f;
    ((float4*)&f)[0] = *((const float4*)p);
    ((float4*)&f)[1] = *((const float4*)(p + 16));
    return f;
}

// ---------------- fp32 -> f16 converts ----------------
__global__ void attn66_cvt_x(const float* __restrict__ x, _Float16* __restrict__ xh, int n) {
    int i = blockIdx.x * blockDim.x + threadIdx.x;
    if (i < n) xh[i] = (_Float16)x[i];
}

// W [DIM][QKVC] fp32  ->  whT [QKVC][DIM] f16
__global__ void attn66_cvt_wT(const float* __restrict__ W, _Float16* __restrict__ whT) {
    int i = blockIdx.x * blockDim.x + threadIdx.x;
    if (i < DIM * QKVC) {
        int k = i / QKVC;
        int c = i % QKVC;
        whT[(size_t)c * DIM + k] = (_Float16)W[i];
    }
}

// ---------------- QKV projection GEMM ----------------
// C[16384 x 2304] = Xh[16384 x 768] * WhT^T + bias.
// 64x64 block tile, 4 waves, each wave: 16 rows x 64 cols (4 WMMA C tiles).
__global__ __launch_bounds__(128)
void attn66_qkv_gemm(const _Float16* __restrict__ xh,
                     const _Float16* __restrict__ whT,
                     const float*    __restrict__ bias,
                     _Float16* __restrict__ qh,
                     _Float16* __restrict__ kh,
                     _Float16* __restrict__ vT) {
    __shared__ __align__(16) _Float16 As[64][32];
    __shared__ __align__(16) _Float16 Bs[64][32];

    const int t    = threadIdx.x;
    const int wave = t >> 5;
    const int lane = t & 31;
    const int lm   = lane & 15;
    const int lk8  = (lane >> 4) << 3;
    const int m0   = blockIdx.y * 64;
    const int n0   = blockIdx.x * 64;

    v8f acc[4] = {};

    for (int k0 = 0; k0 < DIM; k0 += 32) {
        #pragma unroll
        for (int cc = 0; cc < 2; ++cc) {
            int ch  = t + cc * 128;       // 0..255
            int row = ch >> 2;            // 0..63
            int kc  = (ch & 3) << 3;      // 0,8,16,24
            *(float4*)&As[row][kc] = *(const float4*)&xh [(size_t)(m0 + row) * DIM + k0 + kc];
            *(float4*)&Bs[row][kc] = *(const float4*)&whT[(size_t)(n0 + row) * DIM + k0 + kc];
        }
        __syncthreads();

        v16h a = load_frag16(&As[wave * 16 + lm][lk8]);
        #pragma unroll
        for (int j = 0; j < 4; ++j) {
            v16h b = load_frag16(&Bs[j * 16 + lm][lk8]);
            acc[j] = __builtin_amdgcn_wmma_f32_16x16x32_f16(
                         false, a, false, b, (short)0, acc[j], false, false);
        }
        __syncthreads();
    }

    const int rbase = m0 + wave * 16 + lk8;
    #pragma unroll
    for (int j = 0; j < 4; ++j) {
        const int   col = n0 + j * 16 + lm;
        const float bv  = bias[col];
        #pragma unroll
        for (int r = 0; r < 8; ++r) {
            const int   row = rbase + r;
            const float v   = acc[j][r] + bv;
            if (col < DIM) {
                qh[(size_t)row * DIM + col] = (_Float16)v;
            } else if (col < 2 * DIM) {
                kh[(size_t)row * DIM + (col - DIM)] = (_Float16)v;
            } else {
                const int b   = row >> 12;            // row / 4096
                const int tok = row & (NTOK - 1);
                vT[((size_t)b * DIM + (col - 2 * DIM)) * NTOK + tok] = (_Float16)v;
            }
        }
    }
}

// ---------------- Flash attention ----------------
// Workgroup: 8 waves, 32 query rows, one batch, 64-key blocks.
// Phase 1: wave w computes ONE full-depth 16x16 S tile (Mtile = w>>2, keytile = w&3):
//          24 chained WMMAs over K=768, plain ds_store of the result (no atomics).
// Phase 2: all 256 threads run the shared online softmax: 8 lanes per query row,
//          __shfl_xor(width=8) reductions for row max / row sum.
// Phase 3: wave w owns head-dim slice [96w,96w+96): rescale O by alpha, O += P*V via WMMA.
// Q tile staged in LDS once via TDM tensor_load_to_lds; K and V^T fragments stream
// straight from global memory (all of K/V lives in the 192MB L2), with full-block
// global_prefetch warming of the next K/V tiles.
__global__ __launch_bounds__(256)
void attn66_attention(const _Float16* __restrict__ qh,
                      const _Float16* __restrict__ kh,
                      const _Float16* __restrict__ vT,
                      float* __restrict__ out) {
    __shared__ __align__(16) _Float16 Qs[32][DIM];   // 48KB
    __shared__ __align__(16) float    Sb[32][KB];    // 8KB
    __shared__ __align__(16) _Float16 Pb[32][KB];    // 4KB
    __shared__ float mrow[32], lrow[32], arow[32];

    const int t    = threadIdx.x;
    const int wave = t >> 5;
    const int lane = t & 31;
    const int lm   = lane & 15;
    const int lk8  = (lane >> 4) << 3;
    const int b    = blockIdx.y;
    const int q0   = blockIdx.x * 32;
    const int mi   = wave >> 2;      // phase-1 M tile
    const int kj   = wave & 3;       // phase-1 key tile
    const int ds   = wave * 96;      // phase-3 head-dim slice
    const float scale = 0.03608439182435161f;   // 1/sqrt(768)

    const _Float16* qb  = qh + ((size_t)b * NTOK + q0) * DIM;
    const _Float16* kbp = kh + (size_t)b * NTOK * DIM;
    const _Float16* vbp = vT + (size_t)b * DIM * NTOK;

    if (t < 32) { mrow[t] = -1e30f; lrow[t] = 0.f; }

    // ---- stage Q (32 x 768 f16) into LDS ----
#if HAVE_TDM
    if (t < 32) {   // wave 0 issues one TDM descriptor for the whole tile
        attn66_tdm_load_2d_f16(attn66_lds_offset(&Qs[0][0]), qb,
                               /*tensor_d0=*/DIM, /*tensor_d1=*/32,
                               /*tile_d0=*/DIM, /*tile_d1=*/32,
                               /*stride0=*/DIM);
        __builtin_amdgcn_s_wait_tensorcnt(0);
    }
#else
    for (int idx = t; idx < 32 * (DIM / 8); idx += 256) {
        int row = idx / (DIM / 8);
        int c8  = (idx % (DIM / 8)) * 8;
        *(float4*)&Qs[row][c8] = *(const float4*)&qb[(size_t)row * DIM + c8];
    }
#endif
    __syncthreads();

    v8f o[2][6] = {};

    #pragma unroll 1
    for (int kblk = 0; kblk < NTOK; kblk += KB) {
        // Warm L2->WGP path for the next K block (96KB = 768 lines) and the
        // next V^T columns (768 rows x 128B) while the matrix cores chew on this one.
        if (kblk + KB < NTOK) {
            const char* kn = (const char*)(kbp + (size_t)(kblk + KB) * DIM);
            const char* vn = (const char*)(vbp + kblk + KB);
            #pragma unroll
            for (int pp = 0; pp < 3; ++pp) {
                __builtin_prefetch(kn + (size_t)(t + pp * 256) * 128, 0, 1);
                __builtin_prefetch(vn + (size_t)(t + pp * 256) * (NTOK * 2), 0, 1);
            }
        }

        // ---- Phase 1: S tile = Q[mi] * K[kj]^T over full 768 depth ----
        {
            v8f s = {};
            const _Float16* qrow = &Qs[mi * 16 + lm][lk8];
            const _Float16* krow = kbp + (size_t)(kblk + kj * 16 + lm) * DIM + lk8;
            #pragma unroll
            for (int ks = 0; ks < DIM / 32; ++ks) {
                v16h a  = load_frag16(qrow + ks * 32);
                v16h kf = load_frag16(krow + ks * 32);
                s = __builtin_amdgcn_wmma_f32_16x16x32_f16(
                        false, a, false, kf, (short)0, s, false, false);
            }
            const int srow = mi * 16 + lk8;
            const int scol = kj * 16 + lm;
            #pragma unroll
            for (int r = 0; r < 8; ++r) Sb[srow + r][scol] = s[r];
        }
        __syncthreads();

        // ---- Phase 2: shared online softmax, 8 lanes per query row ----
        {
            const int row = t >> 3;        // 0..31
            const int sub = t & 7;         // 8 lanes per row
            const int c0  = sub * 8;       // 8 columns each
            float mo = mrow[row];
            float mx = -1e30f;
            #pragma unroll
            for (int c = 0; c < 8; ++c) mx = fmaxf(mx, Sb[row][c0 + c] * scale);
            #pragma unroll
            for (int sh = 1; sh < 8; sh <<= 1) mx = fmaxf(mx, __shfl_xor(mx, sh, 8));
            mx = fmaxf(mx, mo);
            float ladd = 0.f;
            #pragma unroll
            for (int c = 0; c < 8; ++c) {
                float e = __expf(Sb[row][c0 + c] * scale - mx);
                Pb[row][c0 + c] = (_Float16)e;
                ladd += e;
            }
            #pragma unroll
            for (int sh = 1; sh < 8; sh <<= 1) ladd += __shfl_xor(ladd, sh, 8);
            if (sub == 0) {
                float alpha = __expf(mo - mx);
                lrow[row] = lrow[row] * alpha + ladd;
                mrow[row] = mx;
                arow[row] = alpha;
            }
        }
        __syncthreads();

        // ---- Phase 3: O = alpha*O + P * V_slice ----
        #pragma unroll
        for (int i = 0; i < 2; ++i) {
            float al[8];
            #pragma unroll
            for (int r = 0; r < 8; ++r) al[r] = arow[i * 16 + lk8 + r];
            #pragma unroll
            for (int nj = 0; nj < 6; ++nj)
                #pragma unroll
                for (int r = 0; r < 8; ++r) o[i][nj][r] *= al[r];

            #pragma unroll
            for (int ks = 0; ks < KB / 32; ++ks) {
                v16h pf = load_frag16(&Pb[i * 16 + lm][ks * 32 + lk8]);
                #pragma unroll
                for (int nj = 0; nj < 6; ++nj) {
                    v16h vf = load_frag16(vbp + (size_t)(ds + nj * 16 + lm) * NTOK
                                              + kblk + ks * 32 + lk8);
                    o[i][nj] = __builtin_amdgcn_wmma_f32_16x16x32_f16(
                                   false, pf, false, vf, (short)0, o[i][nj], false, false);
                }
            }
        }
        __syncthreads();
    }

    // ---- Epilogue: normalize by l, store fp32 ----
    float* ob = out + ((size_t)b * NTOK + q0) * DIM;
    #pragma unroll
    for (int i = 0; i < 2; ++i) {
        float linv[8];
        #pragma unroll
        for (int r = 0; r < 8; ++r) linv[r] = 1.f / lrow[i * 16 + lk8 + r];
        #pragma unroll
        for (int nj = 0; nj < 6; ++nj) {
            const int col = ds + nj * 16 + lm;
            #pragma unroll
            for (int r = 0; r < 8; ++r)
                ob[(size_t)(i * 16 + lk8 + r) * DIM + col] = o[i][nj][r] * linv[r];
        }
    }
}

// ---------------- host launcher ----------------
extern "C" void kernel_launch(void* const* d_in, const int* in_sizes, int n_in,
                              void* d_out, int out_size, void* d_ws, size_t ws_size,
                              hipStream_t stream) {
    const float* x    = (const float*)d_in[0];   // [4,4096,768]
    const float* W    = (const float*)d_in[1];   // [768,2304]
    const float* bias = (const float*)d_in[2];   // [2304]
    float*       out  = (float*)d_out;           // [4,4096,768]

    char* ws = (char*)d_ws;
    size_t off = 0;
    _Float16* xh  = (_Float16*)(ws + off); off += (size_t)NB * NTOK * DIM * 2;  // 24 MB
    _Float16* whT = (_Float16*)(ws + off); off += (size_t)QKVC * DIM * 2;       // 3.4 MB
    _Float16* qh  = (_Float16*)(ws + off); off += (size_t)NB * NTOK * DIM * 2;  // 24 MB
    _Float16* kh  = (_Float16*)(ws + off); off += (size_t)NB * NTOK * DIM * 2;  // 24 MB
    _Float16* vT  = (_Float16*)(ws + off); off += (size_t)NB * NTOK * DIM * 2;  // 24 MB

    {
        int n = NB * NTOK * DIM;
        attn66_cvt_x<<<(n + 255) / 256, 256, 0, stream>>>(x, xh, n);
    }
    attn66_cvt_wT<<<(DIM * QKVC + 255) / 256, 256, 0, stream>>>(W, whT);

    attn66_qkv_gemm<<<dim3(QKVC / 64, (NB * NTOK) / 64), 128, 0, stream>>>(
        xh, whT, bias, qh, kh, vT);

    attn66_attention<<<dim3(NTOK / 32, NB), 256, 0, stream>>>(qh, kh, vT, out);
}
